// RowColRPA_20237885899403
// MI455X (gfx1250) — matrix operand order repaired and checked by reading, not compiled, not run on hardware
//
#include <hip/hip_runtime.h>

// ---------------------------------------------------------------------------
// RowColRPA block for MI455X (gfx1250, wave32, WMMA).
// All five 256x256 GEMMs + both attention einsums run on v_wmma_f32_16x16x32_f16.
// ---------------------------------------------------------------------------

typedef _Float16 v16h __attribute__((ext_vector_type(16)));
typedef _Float16 v8h  __attribute__((ext_vector_type(8)));
typedef float    v8f  __attribute__((ext_vector_type(8)));

#define BATCH   2
#define SEQ     512
#define DIM     256
#define HEADS   8
#define HDIM    32
#define TOKENS  (BATCH * SEQ)       // 1024
#define LN_EPS  1e-5f

// ---- WMMA fragment loaders (A/B in contiguous k order, wave32) --------------

// A-matrix 16x32 f16 (MxK). Lane L: m = m0 + (L&15); half = L>>4.
// elem e -> K = k0 + 16*half + e (matches load_b32x16).
// Both chunks are contiguous 16-byte loads from row-major A (lda in halves).
__device__ __forceinline__ v16h load_a16x32(const _Float16* A, int lda, int m0, int k0) {
    const int lane = threadIdx.x & 31;
    const int half = lane >> 4;
    const int m    = m0 + (lane & 15);
    const _Float16* p = A + (size_t)m * lda + k0 + half * 16;
    v8h lo = *(const v8h*)p;
    v8h hi = *(const v8h*)(p + 8);
    return __builtin_shufflevector(lo, hi, 0,1,2,3,4,5,6,7,8,9,10,11,12,13,14,15);
}

// B-matrix 32x16 f16 (KxN) where B[k][n] = Bsrc[n0+n][k0+k] (row-major over k,
// i.e. the "NT" weight layout W[n, k]). Lane L: n = n0 + (L&15); khalf = L>>4;
// elem e -> K = k0 + 16*khalf + e  => one contiguous 32-byte load per lane.
__device__ __forceinline__ v16h load_b32x16(const _Float16* B, int ldb, int n0, int k0) {
    const int lane  = threadIdx.x & 31;
    const int khalf = lane >> 4;
    const int n     = n0 + (lane & 15);
    return *(const v16h*)(B + (size_t)n * ldb + k0 + khalf * 16);
}

__device__ __forceinline__ v8f wmma_f16(v16h a, v16h b, v8f c) {
    return __builtin_amdgcn_wmma_f32_16x16x32_f16(
        /*neg_a=*/false, a, /*neg_b=*/false, b,
        /*c_mod=*/(short)0, c, /*reuse_a=*/false, /*reuse_b=*/false);
}

// C/D 16x16 f32 layout: lane L, vgpr j -> row m0 + 8*(L>>4) + j, col n0 + (L&15)
__device__ __forceinline__ void store_tile(float* C, int ldc, int m0, int n0,
                                           v8f c, const float* bias) {
    const int lane = threadIdx.x & 31;
    const int half = lane >> 4;
    const int n    = lane & 15;
    const float bv = bias ? bias[n0 + n] : 0.0f;
#pragma unroll
    for (int j = 0; j < 8; ++j)
        C[(size_t)(m0 + half * 8 + j) * ldc + n0 + n] = c[j] + bv;
}

// ---- small helper kernels --------------------------------------------------

__global__ void cvt_f32_f16(const float* __restrict__ in, _Float16* __restrict__ out, int n) {
    for (int i = blockIdx.x * blockDim.x + threadIdx.x; i < n; i += gridDim.x * blockDim.x)
        out[i] = (_Float16)in[i];
}

// P min/max over tokens per batch & channel.  grid = (BATCH), block = DIM.
__global__ void pminmax_kernel(const float* __restrict__ Pf,
                               float* __restrict__ pmax, float* __restrict__ pmin) {
    const int b = blockIdx.x, e = threadIdx.x;
    float mx = -3.402823466e38f, mn = 3.402823466e38f;
    for (int n = 0; n < SEQ; ++n) {
        float v = Pf[(size_t)(b * SEQ + n) * DIM + e];
        mx = fmaxf(mx, v);
        mn = fminf(mn, v);
    }
    pmax[b * DIM + e] = mx;
    pmin[b * DIM + e] = mn;
}

// Qr = Q+bq+pmax-P+br ; Kr = K+bk+P-pmin+br ; V = V+bv, emitted in f16
// attention layouts: Qr/Kr -> [b,h,n,32] ; V -> transposed [b,h,32,512].
__global__ void combine_kernel(const float* __restrict__ Qf, const float* __restrict__ Kf,
                               const float* __restrict__ Vf, const float* __restrict__ Pf,
                               const float* __restrict__ bq, const float* __restrict__ bk,
                               const float* __restrict__ bv, const float* __restrict__ br,
                               const float* __restrict__ pmax, const float* __restrict__ pmin,
                               _Float16* __restrict__ Qrh, _Float16* __restrict__ Krh,
                               _Float16* __restrict__ Vth) {
    const int bn = blockIdx.x, e = threadIdx.x;
    const int b = bn >> 9, n = bn & (SEQ - 1);
    const size_t i = (size_t)bn * DIM + e;
    const float p = Pf[i];
    const float q = Qf[i] + bq[e] + pmax[b * DIM + e] - p + br[e];
    const float k = Kf[i] + bk[e] + p - pmin[b * DIM + e] + br[e];
    const float v = Vf[i] + bv[e];
    const int h = e >> 5, d = e & 31;
    const size_t bh = (size_t)(b * HEADS + h);
    Qrh[(bh * SEQ + n) * HDIM + d] = (_Float16)q;
    Krh[(bh * SEQ + n) * HDIM + d] = (_Float16)k;
    Vth[(bh * HDIM + d) * SEQ + n] = (_Float16)v;
}

// Fused Q/K/V/P projection GEMM: out = xh @ W^T  (bias handled later).
// One wave computes a 32x32 tile; 1024 tiles total -> 128 blocks x 8 waves.
__global__ void gemm_qkvp(const _Float16* __restrict__ xh,
                          const _Float16* __restrict__ wq, const _Float16* __restrict__ wk,
                          const _Float16* __restrict__ wv, const _Float16* __restrict__ wr,
                          float* __restrict__ Qf, float* __restrict__ Kf,
                          float* __restrict__ Vf, float* __restrict__ Pf) {
    const int tile = blockIdx.x * 8 + (threadIdx.x >> 5);   // 0..1023
    const int mat  = tile >> 8;                             // 256 tiles per matrix
    const int rem  = tile & 255;
    const int m0   = (rem >> 3) * 32;                       // 32 m-tiles
    const int n0   = (rem & 7) * 32;                        // 8 n-tiles
    const _Float16* W = (mat == 0) ? wq : (mat == 1) ? wk : (mat == 2) ? wv : wr;
    float*          O = (mat == 0) ? Qf : (mat == 1) ? Kf : (mat == 2) ? Vf : Pf;

    v8f c00 = {}, c01 = {}, c10 = {}, c11 = {};
#pragma unroll
    for (int k0 = 0; k0 < DIM; k0 += 32) {
        v16h a0 = load_a16x32(xh, DIM, m0,      k0);
        v16h a1 = load_a16x32(xh, DIM, m0 + 16, k0);
        v16h b0 = load_b32x16(W,  DIM, n0,      k0);
        v16h b1 = load_b32x16(W,  DIM, n0 + 16, k0);
        c00 = wmma_f16(a0, b0, c00);
        c01 = wmma_f16(a0, b1, c01);
        c10 = wmma_f16(a1, b0, c10);
        c11 = wmma_f16(a1, b1, c11);
    }
    store_tile(O, DIM, m0,      n0,      c00, nullptr);
    store_tile(O, DIM, m0,      n0 + 16, c01, nullptr);
    store_tile(O, DIM, m0 + 16, n0,      c10, nullptr);
    store_tile(O, DIM, m0 + 16, n0 + 16, c11, nullptr);
}

// Attention: one wave per (b, h, 16-row q tile). scores = Qr Kr^T / sqrt(32),
// softmax over 512 keys, attn = probs @ V. 64 WMMA per wave.
__global__ void attention_kernel(const _Float16* __restrict__ Qrh,
                                 const _Float16* __restrict__ Krh,
                                 const _Float16* __restrict__ Vth,
                                 _Float16* __restrict__ attnh) {
    __shared__ __align__(32) float    s_scores[16][SEQ];   // 32 KB
    __shared__ __align__(32) _Float16 s_probs[16][SEQ];    // 16 KB

    const int idx = blockIdx.x;                 // 0..511
    const int b   = idx >> 8;
    const int h   = (idx >> 5) & (HEADS - 1);
    const int q0  = (idx & 31) * 16;
    const int lane = threadIdx.x & 31;
    const int half = lane >> 4, nn = lane & 15;

    const size_t bh = (size_t)(b * HEADS + h);
    const _Float16* Qbh = Qrh + bh * SEQ * HDIM;
    const _Float16* Kbh = Krh + bh * SEQ * HDIM;
    const _Float16* Vbh = Vth + bh * HDIM * SEQ;
    const float scale = 0.17677669529663687f;   // 1/sqrt(32)

    // ---- scores: 32 WMMA (Dh=32 => single k-step each) ----
    v16h aQ = load_a16x32(Qbh, HDIM, q0, 0);
#pragma unroll 4
    for (int kt = 0; kt < 32; ++kt) {
        v16h bK = load_b32x16(Kbh, HDIM, kt * 16, 0);
        v8f s = {};
        s = wmma_f16(aQ, bK, s);
#pragma unroll
        for (int j = 0; j < 8; ++j)
            s_scores[half * 8 + j][kt * 16 + nn] = s[j] * scale;
    }
    __syncthreads();

    // ---- softmax over 512 keys (lanes 0..15 own one row each) ----
    if (lane < 16) {
        const int r = lane;
        float mx = -3.402823466e38f;
        for (int k = 0; k < SEQ; ++k) mx = fmaxf(mx, s_scores[r][k]);
        float sum = 0.0f;
        for (int k = 0; k < SEQ; ++k) {
            float e = __expf(s_scores[r][k] - mx);
            sum += e;
            s_scores[r][k] = e;
        }
        const float inv = 1.0f / sum;
        for (int k = 0; k < SEQ; ++k)
            s_probs[r][k] = (_Float16)(s_scores[r][k] * inv);
    }
    __syncthreads();

    // ---- attn = probs @ V : 16 k-steps x 2 d-tiles = 32 WMMA ----
    v8f c0 = {}, c1 = {};
#pragma unroll 4
    for (int kt = 0; kt < 16; ++kt) {
        v16h aP  = load_a16x32(&s_probs[0][0], SEQ, 0, kt * 32);
        v16h bV0 = load_b32x16(Vbh, SEQ, 0,  kt * 32);
        v16h bV1 = load_b32x16(Vbh, SEQ, 16, kt * 32);
        c0 = wmma_f16(aP, bV0, c0);
        c1 = wmma_f16(aP, bV1, c1);
    }

    // write attn into [token, 256] f16 for the output-projection GEMM
#pragma unroll
    for (int j = 0; j < 8; ++j) {
        const size_t row = (size_t)b * SEQ + q0 + half * 8 + j;
        attnh[row * DIM + h * HDIM + nn]      = (_Float16)c0[j];
        attnh[row * DIM + h * HDIM + 16 + nn] = (_Float16)c1[j];
    }
}

// Output projection: y = attnh @ Wo^T + bo.  256 tiles -> 32 blocks x 8 waves.
__global__ void gemm_oproj(const _Float16* __restrict__ attnh,
                           const _Float16* __restrict__ wo,
                           const float* __restrict__ bo, float* __restrict__ y) {
    const int tile = blockIdx.x * 8 + (threadIdx.x >> 5);   // 0..255
    const int m0 = (tile >> 3) * 32;
    const int n0 = (tile & 7) * 32;
    v8f c00 = {}, c01 = {}, c10 = {}, c11 = {};
#pragma unroll
    for (int k0 = 0; k0 < DIM; k0 += 32) {
        v16h a0 = load_a16x32(attnh, DIM, m0,      k0);
        v16h a1 = load_a16x32(attnh, DIM, m0 + 16, k0);
        v16h b0 = load_b32x16(wo,    DIM, n0,      k0);
        v16h b1 = load_b32x16(wo,    DIM, n0 + 16, k0);
        c00 = wmma_f16(a0, b0, c00);
        c01 = wmma_f16(a0, b1, c01);
        c10 = wmma_f16(a1, b0, c10);
        c11 = wmma_f16(a1, b1, c11);
    }
    store_tile(y, DIM, m0,      n0,      c00, bo);
    store_tile(y, DIM, m0,      n0 + 16, c01, bo);
    store_tile(y, DIM, m0 + 16, n0,      c10, bo);
    store_tile(y, DIM, m0 + 16, n0 + 16, c11, bo);
}

// residual + LayerNorm + LayerNorm, one 256-thread block per token.
__global__ void resid_ln2_kernel(const float* __restrict__ x, const float* __restrict__ y,
                                 const float* __restrict__ g1, const float* __restrict__ b1,
                                 const float* __restrict__ g2, const float* __restrict__ b2,
                                 float* __restrict__ out) {
    __shared__ float red[DIM];
    const int bn = blockIdx.x, t = threadIdx.x;
    const size_t i = (size_t)bn * DIM + t;
    const float z = x[i] + y[i];

    red[t] = z; __syncthreads();
    for (int s = DIM / 2; s > 0; s >>= 1) { if (t < s) red[t] += red[t + s]; __syncthreads(); }
    const float mu = red[0] * (1.0f / DIM); __syncthreads();
    const float zc = z - mu;
    red[t] = zc * zc; __syncthreads();
    for (int s = DIM / 2; s > 0; s >>= 1) { if (t < s) red[t] += red[t + s]; __syncthreads(); }
    const float var = red[0] * (1.0f / DIM); __syncthreads();
    const float hv = zc * rsqrtf(var + LN_EPS) * g1[t] + b1[t];

    red[t] = hv; __syncthreads();
    for (int s = DIM / 2; s > 0; s >>= 1) { if (t < s) red[t] += red[t + s]; __syncthreads(); }
    const float mu2 = red[0] * (1.0f / DIM); __syncthreads();
    const float hc = hv - mu2;
    red[t] = hc * hc; __syncthreads();
    for (int s = DIM / 2; s > 0; s >>= 1) { if (t < s) red[t] += red[t + s]; __syncthreads(); }
    const float var2 = red[0] * (1.0f / DIM);
    out[i] = hc * rsqrtf(var2 + LN_EPS) * g2[t] + b2[t];
}

// ---------------------------------------------------------------------------

extern "C" void kernel_launch(void* const* d_in, const int* in_sizes, int n_in,
                              void* d_out, int out_size, void* d_ws, size_t ws_size,
                              hipStream_t stream) {
    (void)in_sizes; (void)n_in; (void)out_size; (void)ws_size;

    const float* x  = (const float*)d_in[0];
    const float* Wq = (const float*)d_in[1];  const float* bq = (const float*)d_in[2];
    const float* Wk = (const float*)d_in[3];  const float* bk = (const float*)d_in[4];
    const float* Wv = (const float*)d_in[5];  const float* bv = (const float*)d_in[6];
    const float* Wr = (const float*)d_in[7];  const float* br = (const float*)d_in[8];
    const float* Wo = (const float*)d_in[9];  const float* bo = (const float*)d_in[10];
    const float* g1 = (const float*)d_in[11]; const float* b1 = (const float*)d_in[12];
    const float* g2 = (const float*)d_in[13]; const float* b2 = (const float*)d_in[14];
    float* out = (float*)d_out;

    // workspace carve-up (256B aligned)
    char* ws = (char*)d_ws;
    size_t off = 0;
    auto alloc = [&](size_t bytes) -> void* {
        void* p = ws + off;
        off += (bytes + 255) & ~(size_t)255;
        return p;
    };
    _Float16* xh    = (_Float16*)alloc((size_t)TOKENS * DIM * 2);
    _Float16* whq   = (_Float16*)alloc((size_t)DIM * DIM * 2);
    _Float16* whk   = (_Float16*)alloc((size_t)DIM * DIM * 2);
    _Float16* whv   = (_Float16*)alloc((size_t)DIM * DIM * 2);
    _Float16* whr   = (_Float16*)alloc((size_t)DIM * DIM * 2);
    _Float16* who   = (_Float16*)alloc((size_t)DIM * DIM * 2);
    float*    Qf    = (float*)alloc((size_t)TOKENS * DIM * 4);
    float*    Kf    = (float*)alloc((size_t)TOKENS * DIM * 4);
    float*    Vf    = (float*)alloc((size_t)TOKENS * DIM * 4);
    float*    Pf    = (float*)alloc((size_t)TOKENS * DIM * 4);
    float*    pmax  = (float*)alloc((size_t)BATCH * DIM * 4);
    float*    pmin  = (float*)alloc((size_t)BATCH * DIM * 4);
    _Float16* Qrh   = (_Float16*)alloc((size_t)TOKENS * DIM * 2);
    _Float16* Krh   = (_Float16*)alloc((size_t)TOKENS * DIM * 2);
    _Float16* Vth   = (_Float16*)alloc((size_t)TOKENS * DIM * 2);
    _Float16* attnh = (_Float16*)alloc((size_t)TOKENS * DIM * 2);
    float*    yproj = (float*)alloc((size_t)TOKENS * DIM * 4);

    // 1) f32 -> f16 converts
    cvt_f32_f16<<<256, 256, 0, stream>>>(x,  xh,  TOKENS * DIM);
    cvt_f32_f16<<<64,  256, 0, stream>>>(Wq, whq, DIM * DIM);
    cvt_f32_f16<<<64,  256, 0, stream>>>(Wk, whk, DIM * DIM);
    cvt_f32_f16<<<64,  256, 0, stream>>>(Wv, whv, DIM * DIM);
    cvt_f32_f16<<<64,  256, 0, stream>>>(Wr, whr, DIM * DIM);
    cvt_f32_f16<<<64,  256, 0, stream>>>(Wo, who, DIM * DIM);

    // 2) fused Q/K/V/P projection (WMMA)
    gemm_qkvp<<<128, 256, 0, stream>>>(xh, whq, whk, whv, whr, Qf, Kf, Vf, Pf);

    // 3) per-batch channel min/max of P (collapses the O(N^2) pairwise max)
    pminmax_kernel<<<BATCH, DIM, 0, stream>>>(Pf, pmax, pmin);

    // 4) bias/bias-combine + layout for attention
    combine_kernel<<<TOKENS, DIM, 0, stream>>>(Qf, Kf, Vf, Pf, bq, bk, bv, br,
                                               pmax, pmin, Qrh, Krh, Vth);

    // 5) attention (WMMA QK^T + softmax + WMMA PV)
    attention_kernel<<<BATCH * HEADS * (SEQ / 16), 32, 0, stream>>>(Qrh, Krh, Vth, attnh);

    // 6) output projection (WMMA) + bias
    gemm_oproj<<<32, 256, 0, stream>>>(attnh, who, bo, yproj);

    // 7) residual + double LayerNorm
    resid_ln2_kernel<<<TOKENS, DIM, 0, stream>>>(x, yproj, g1, b1, g2, b2, out);
}
